// SVDLinearModule_88914412962285
// MI455X (gfx1250) — matrix operand order repaired and checked by reading, not compiled, not run on hardware
//
#include <hip/hip_runtime.h>

// ---------------------------------------------------------------------------
// SVD-factored linear:  out = ((x @ V) * S) @ U^T + bias
// Stage 1: Y[m,r] = (sum_k x[m,k]*V[k,r]) * S[r]      (bf16 in d_ws, 16 MB)
// Stage 2: out[m,i] = sum_r Y[m,r]*U[i,r] + bias[i]
// v_wmma_f32_16x16x32_bf16, fp32 accumulate, double-buffered LDS (BK=64),
// async global->LDS (ASYNCcnt) for the bf16 Y operand in stage 2.
// ---------------------------------------------------------------------------

typedef __attribute__((ext_vector_type(16))) __bf16 v16bf;
typedef __attribute__((ext_vector_type(8)))  __bf16 v8bf;
typedef __attribute__((ext_vector_type(4)))  __bf16 v4bf;
typedef __attribute__((ext_vector_type(8)))  float  v8f;

#define M_TOT 8192
#define K1    4096   // OUT_H (stage-1 reduction dim)
#define N1    1024   // RANK
#define K2    1024   // RANK (stage-2 reduction dim)
#define N2    4096   // IN_H

#define BM 128
#define BN 128
#define BK 64
#define LDSS 80      // bf16 elems per LDS row: 160B stride (16B aligned)

#define NIT1 (K1 / BK)   // 64
#define NIT2 (K2 / BK)   // 16

// ---- CDNA5 async global->LDS (ISA 10.x / §15.18.3, GVS addressing) --------

__device__ __forceinline__ void async_load_b128(unsigned lds_addr,
                                                unsigned goff_bytes,
                                                unsigned long long gbase) {
  asm volatile("global_load_async_to_lds_b128 %0, %1, %2 offset:0"
               :
               : "v"(lds_addr), "v"(goff_bytes), "s"(gbase)
               : "memory");
}

__device__ __forceinline__ void wait_asynccnt0() {
#if __has_builtin(__builtin_amdgcn_s_wait_asynccnt)
  __builtin_amdgcn_s_wait_asynccnt(0);
#else
  asm volatile("s_wait_asynccnt 0x0" ::: "memory");
#endif
}

// ---- WMMA fragment gathers from LDS ---------------------------------------

__device__ __forceinline__ v16bf load_a_frag(const __bf16* ap) {
  // ISA 7.12.2 16-bit A: lanes 0-15 K{0..7,16..23}, lanes 16-31 K{8..15,24..31}
  v8bf a0 = *reinterpret_cast<const v8bf*>(ap);
  v8bf a1 = *reinterpret_cast<const v8bf*>(ap + 16);
  v16bf a;
#pragma unroll
  for (int e = 0; e < 8; ++e) { a[e] = a0[e]; a[e + 8] = a1[e]; }
  return a;
}

__device__ __forceinline__ v16bf load_b_frag(const __bf16* bp) {
  // B stored N-major in LDS: lane's column, K{0..7,16..23} / K{8..15,24..31}
  v8bf b0 = *reinterpret_cast<const v8bf*>(bp);
  v8bf b1 = *reinterpret_cast<const v8bf*>(bp + 16);
  v16bf b;
#pragma unroll
  for (int e = 0; e < 8; ++e) { b[e] = b0[e]; b[e + 8] = b1[e]; }
  return b;
}

__device__ __forceinline__ v8bf cvt8(const float4& a, const float4& b) {
  v8bf r;
  r[0] = (__bf16)a.x; r[1] = (__bf16)a.y; r[2] = (__bf16)a.z; r[3] = (__bf16)a.w;
  r[4] = (__bf16)b.x; r[5] = (__bf16)b.y; r[6] = (__bf16)b.z; r[7] = (__bf16)b.w;
  return r;
}

// ---- Stage 1: Y = (x @ V) * S ---------------------------------------------

__global__ __launch_bounds__(256) void svd_stage1(
    const float* __restrict__ X, const float* __restrict__ V,
    const float* __restrict__ S, __bf16* __restrict__ Y)
{
  __shared__ __bf16 As[2][BM][LDSS];   // x tile, M-major, K contiguous
  __shared__ __bf16 Bs[2][BN][LDSS];   // V tile, N-major, K contiguous

  const int t     = threadIdx.x;
  const int lane  = t & 31;
  const int wave  = t >> 5;
  const int waveM = wave >> 1;      // 0..3 : 32-row slabs
  const int waveN = wave & 1;       // 0..1 : 64-col slabs
  const int lh    = lane & 15;
  const int ls    = lane >> 4;

  const int mbase = blockIdx.y * BM;
  const int nbase = blockIdx.x * BN;

  // x tile 128x64 f32: thread owns 8 contiguous K -> one b128 LDS store/task
  auto load_x = [&](int kb, float4* pa) {
#pragma unroll
    for (int i = 0; i < 4; ++i) {
      int idx = t + i * 256;
      int row = idx >> 3, c8 = (idx & 7) * 8;
      const float* p = &X[(size_t)(mbase + row) * K1 + kb + c8];
      pa[2 * i]     = *reinterpret_cast<const float4*>(p);
      pa[2 * i + 1] = *reinterpret_cast<const float4*>(p + 4);
    }
  };
  auto commit_x = [&](int buf, const float4* pa) {
#pragma unroll
    for (int i = 0; i < 4; ++i) {
      int idx = t + i * 256;
      int row = idx >> 3, c8 = (idx & 7) * 8;
      *reinterpret_cast<v8bf*>(&As[buf][row][c8]) =
          cvt8(pa[2 * i], pa[2 * i + 1]);
    }
  };
  // V tile 64x128 f32 -> N-major: 4x4 micro-transpose in registers
  auto load_v = [&](int kb, float4* pb) {
#pragma unroll
    for (int i = 0; i < 2; ++i) {
      int idx = t + i * 256;
      int kg = idx >> 5, ng = idx & 31;     // 16 k-groups x 32 n-groups
#pragma unroll
      for (int r = 0; r < 4; ++r)
        pb[i * 4 + r] = *reinterpret_cast<const float4*>(
            &V[(size_t)(kb + kg * 4 + r) * N1 + nbase + ng * 4]);
    }
  };
  auto commit_v = [&](int buf, const float4* pb) {
#pragma unroll
    for (int i = 0; i < 2; ++i) {
      int idx = t + i * 256;
      int kg = idx >> 5, ng = idx & 31;
      const float4* q = &pb[i * 4];
#pragma unroll
      for (int e = 0; e < 4; ++e) {
        v4bf b;
        b[0] = (__bf16)((const float*)&q[0])[e];
        b[1] = (__bf16)((const float*)&q[1])[e];
        b[2] = (__bf16)((const float*)&q[2])[e];
        b[3] = (__bf16)((const float*)&q[3])[e];
        *reinterpret_cast<v4bf*>(&Bs[buf][ng * 4 + e][kg * 4]) = b;
      }
    }
  };

  v8f acc[2][4];
#pragma unroll
  for (int mt = 0; mt < 2; ++mt)
#pragma unroll
    for (int nt = 0; nt < 4; ++nt) acc[mt][nt] = (v8f){};

  // prologue: chunk 0 -> buffer 0
  {
    float4 pa[8], pb[8];
    load_x(0, pa); load_v(0, pb);
    commit_x(0, pa); commit_v(0, pb);
  }
  __syncthreads();

  for (int it = 0; it < NIT1; ++it) {
    const int cur = it & 1, nxt = cur ^ 1;
    float4 pa[8], pb[8];
    if (it + 1 < NIT1) {           // issue next chunk's global loads early
      load_x((it + 1) * BK, pa);
      load_v((it + 1) * BK, pb);
    }
    // WMMA on current buffer (hides global load latency)
#pragma unroll
    for (int kk = 0; kk < BK; kk += 32) {
      v16bf af[2], bfv[4];
#pragma unroll
      for (int mt = 0; mt < 2; ++mt)
        af[mt] = load_a_frag(&As[cur][waveM * 32 + mt * 16 + lh][kk + ls * 8]);
#pragma unroll
      for (int nt = 0; nt < 4; ++nt)
        bfv[nt] = load_b_frag(&Bs[cur][waveN * 64 + nt * 16 + lh][kk + ls * 8]);
#pragma unroll
      for (int nt = 0; nt < 4; ++nt)
#pragma unroll
        for (int mt = 0; mt < 2; ++mt)
          acc[mt][nt] = __builtin_amdgcn_wmma_f32_16x16x32_bf16(
              false, af[mt], false, bfv[nt], (short)0, acc[mt][nt],
              false, false);
    }
    if (it + 1 < NIT1) {           // convert + commit to the other buffer
      commit_x(nxt, pa);
      commit_v(nxt, pb);
    }
    __syncthreads();
  }

  // epilogue: scale by S[r], store bf16
#pragma unroll
  for (int nt = 0; nt < 4; ++nt) {
    int n = nbase + waveN * 64 + nt * 16 + lh;
    float s = S[n];
#pragma unroll
    for (int mt = 0; mt < 2; ++mt) {
      int m0 = mbase + waveM * 32 + mt * 16 + ls * 8;
#pragma unroll
      for (int j = 0; j < 8; ++j)
        Y[(size_t)(m0 + j) * N1 + n] = (__bf16)(acc[mt][nt][j] * s);
    }
  }
}

// ---- Stage 2: out = Y @ U^T + bias ----------------------------------------

__global__ __launch_bounds__(256) void svd_stage2(
    const __bf16* __restrict__ Yg, const float* __restrict__ U,
    const float* __restrict__ bias, float* __restrict__ out)
{
  __shared__ __bf16 As[2][BM][LDSS];   // Y tile, M-major (filled by async DMA)
  __shared__ __bf16 Bs[2][BN][LDSS];   // U tile: rows of U are K-contiguous

  const int t     = threadIdx.x;
  const int lane  = t & 31;
  const int wave  = t >> 5;
  const int waveM = wave >> 1;
  const int waveN = wave & 1;
  const int lh    = lane & 15;
  const int ls    = lane >> 4;

  const int mbase = blockIdx.y * BM;
  const int nbase = blockIdx.x * BN;

  const unsigned long long ybase = (unsigned long long)(uintptr_t)Yg;

  // Y tile: 128x64 bf16 = 16 KB, 4 async b128 per thread, no VGPR round-trip
  auto issue_async_y = [&](int kb, int buf) {
#pragma unroll
    for (int i = 0; i < 4; ++i) {
      int idx = t + i * 256;
      int row = idx >> 3, c8 = (idx & 7) * 8;
      unsigned lds  = (unsigned)(uintptr_t)&As[buf][row][c8];
      unsigned goff = (unsigned)(((size_t)(mbase + row) * K2 + kb + c8) *
                                 sizeof(__bf16));
      async_load_b128(lds, goff, ybase);
    }
  };
  auto load_u = [&](int kb, float4* pb) {
#pragma unroll
    for (int i = 0; i < 4; ++i) {
      int idx = t + i * 256;
      int n = idx >> 3, c8 = (idx & 7) * 8;
      const float* p = &U[(size_t)(nbase + n) * K2 + kb + c8];
      pb[2 * i]     = *reinterpret_cast<const float4*>(p);
      pb[2 * i + 1] = *reinterpret_cast<const float4*>(p + 4);
    }
  };
  auto commit_u = [&](int buf, const float4* pb) {
#pragma unroll
    for (int i = 0; i < 4; ++i) {
      int idx = t + i * 256;
      int n = idx >> 3, c8 = (idx & 7) * 8;
      *reinterpret_cast<v8bf*>(&Bs[buf][n][c8]) =
          cvt8(pb[2 * i], pb[2 * i + 1]);
    }
  };

  v8f acc[2][4];
#pragma unroll
  for (int mt = 0; mt < 2; ++mt)
#pragma unroll
    for (int nt = 0; nt < 4; ++nt) acc[mt][nt] = (v8f){};

  // prologue: chunk 0 -> buffer 0
  {
    float4 pb[8];
    issue_async_y(0, 0);
    load_u(0, pb);
    commit_u(0, pb);
    wait_asynccnt0();
  }
  __syncthreads();

  for (int it = 0; it < NIT2; ++it) {
    const int cur = it & 1, nxt = cur ^ 1;
    float4 pb[8];
    if (it + 1 < NIT2) {
      issue_async_y((it + 1) * BK, nxt);   // DMA straight into other buffer
      load_u((it + 1) * BK, pb);
    }
#pragma unroll
    for (int kk = 0; kk < BK; kk += 32) {
      v16bf af[2], bfv[4];
#pragma unroll
      for (int mt = 0; mt < 2; ++mt)
        af[mt] = load_a_frag(&As[cur][waveM * 32 + mt * 16 + lh][kk + ls * 8]);
#pragma unroll
      for (int nt = 0; nt < 4; ++nt)
        bfv[nt] = load_b_frag(&Bs[cur][waveN * 64 + nt * 16 + lh][kk + ls * 8]);
#pragma unroll
      for (int nt = 0; nt < 4; ++nt)
#pragma unroll
        for (int mt = 0; mt < 2; ++mt)
          acc[mt][nt] = __builtin_amdgcn_wmma_f32_16x16x32_bf16(
              false, af[mt], false, bfv[nt], (short)0, acc[mt][nt],
              false, false);
    }
    if (it + 1 < NIT2) commit_u(nxt, pb);
    wait_asynccnt0();
    __syncthreads();
  }

  // epilogue: + bias, store fp32
#pragma unroll
  for (int nt = 0; nt < 4; ++nt) {
    int n = nbase + waveN * 64 + nt * 16 + lh;
    float bv = bias[n];
#pragma unroll
    for (int mt = 0; mt < 2; ++mt) {
      int m0 = mbase + waveM * 32 + mt * 16 + ls * 8;
#pragma unroll
      for (int j = 0; j < 8; ++j)
        out[(size_t)(m0 + j) * N2 + n] = acc[mt][nt][j] + bv;
    }
  }
}

// ---------------------------------------------------------------------------

extern "C" void kernel_launch(void* const* d_in, const int* in_sizes, int n_in,
                              void* d_out, int out_size, void* d_ws, size_t ws_size,
                              hipStream_t stream) {
  const float* x    = (const float*)d_in[0];
  const float* U    = (const float*)d_in[1];
  const float* S    = (const float*)d_in[2];
  const float* V    = (const float*)d_in[3];
  const float* bias = (const float*)d_in[4];

  __bf16* Y  = (__bf16*)d_ws;           // 8192*1024 bf16 = 16 MB
  float* out = (float*)d_out;

  svd_stage1<<<dim3(N1 / BN, M_TOT / BM), 256, 0, stream>>>(x, V, S, Y);
  svd_stage2<<<dim3(N2 / BN, M_TOT / BM), 256, 0, stream>>>(Y, U, bias, out);
}
